// GAT_19713899888815
// MI455X (gfx1250) — matrix operand order repaired and checked by reading, not compiled, not run on hardware
//
#include <hip/hip_runtime.h>
#include <hip/hip_bf16.h>

// GAT attention for MI455X (gfx1250, wave32).
// attn_feat matmul done with V_WMMA_F32_16X16X32_F16 (f16 in, f32 accum).
// Softmax math + attn_map output stay in fp32, attn_map stored non-temporal.
// Each block owns 32 i-rows (two 16-wide N-tiles) so every A fragment feeds
// two WMMAs -> halves xT re-read traffic from L2.
//
// WMMA operand layouts used here (A and B share one contiguous k order):
//  A (16x32 f16):  lane m = lane&15; v16h elem e -> K = 16*hi + e
//                  (hi = lane>>4)
//  B (32x16 f16):  lane n = lane&15; v16h elem e -> K = 16*hi + e
//  C/D (16x16 f32): lane n = lane&15; VGPR r -> m = r + 8*hi

typedef __attribute__((ext_vector_type(16))) _Float16 v16h;
typedef __attribute__((ext_vector_type(8)))  _Float16 v8h;
typedef __attribute__((ext_vector_type(8)))  float    v8f;

#define BB 16
#define NN 2048
#define FF 128

__device__ __forceinline__ float lrelu(float s) { return s >= 0.0f ? s : 0.2f * s; }

// ---------------- kernel 1: k = x.wk, q = x.wq ----------------
__global__ __launch_bounds__(256) void kq_kernel(const float* __restrict__ x,
                                                 const float* __restrict__ wk,
                                                 const float* __restrict__ wq,
                                                 float* __restrict__ k,
                                                 float* __restrict__ q) {
    int row = blockIdx.x * 256 + threadIdx.x;          // b*N + n
    const float4* xr  = (const float4*)(x + (size_t)row * FF);
    const float4* wk4 = (const float4*)wk;
    const float4* wq4 = (const float4*)wq;
    float sk = 0.0f, sq = 0.0f;
#pragma unroll 8
    for (int i = 0; i < FF / 4; ++i) {
        float4 v = xr[i];
        float4 a = wk4[i];
        float4 b = wq4[i];
        sk += v.x * a.x + v.y * a.y + v.z * a.z + v.w * a.w;
        sq += v.x * b.x + v.y * b.y + v.z * b.z + v.w * b.w;
    }
    k[row] = sk;
    q[row] = sq;
}

// ---------------- kernel 2: per-batch max of q ----------------
__global__ __launch_bounds__(256) void qmax_kernel(const float* __restrict__ q,
                                                   float* __restrict__ qmax) {
    __shared__ float red[256];
    int b = blockIdx.x, t = threadIdx.x;
    float m = -3.0e38f;
    for (int j = t; j < NN; j += 256) m = fmaxf(m, q[(size_t)b * NN + j]);
    red[t] = m;
    __syncthreads();
    for (int s = 128; s > 0; s >>= 1) {
        if (t < s) red[t] = fmaxf(red[t], red[t + s]);
        __syncthreads();
    }
    if (t == 0) qmax[b] = red[0];
}

// ---------------- kernel 3: invZ[b,i] = 1 / sum_j exp(lrelu(k_i+q_j)-m_i) ----
__global__ __launch_bounds__(256) void invz_kernel(const float* __restrict__ k,
                                                   const float* __restrict__ q,
                                                   const float* __restrict__ qmax,
                                                   float* __restrict__ invz) {
    int wave = threadIdx.x >> 5, lane = threadIdx.x & 31;
    int row  = blockIdx.x * 8 + wave;                  // b*N + i
    int b    = row / NN;
    float ki = k[row];
    float m  = lrelu(ki + qmax[b]);
    const float* qb = q + (size_t)b * NN;
    float acc = 0.0f;
    for (int j = lane; j < NN; j += 32)
        acc += __expf(lrelu(ki + qb[j]) - m);
    for (int off = 16; off > 0; off >>= 1)
        acc += __shfl_down(acc, off, 32);
    if (lane == 0) invz[row] = 1.0f / acc;
}

// ---------------- kernel 4: xT[b,f,n] = (f16) x[b,n,f] ----------------
__global__ __launch_bounds__(256) void xT_kernel(const float* __restrict__ x,
                                                 _Float16* __restrict__ xT) {
    __shared__ _Float16 tile[FF][72];                  // [f][n_local], padded
    int b = blockIdx.y, n0 = blockIdx.x * 64;
    const float* xb = x + ((size_t)b * NN + n0) * FF;
    for (int idx = threadIdx.x; idx < 64 * FF; idx += 256) {
        int nl = idx >> 7, f = idx & 127;              // consecutive t -> consecutive f
        tile[f][nl] = (_Float16)xb[(size_t)nl * FF + f];
    }
    __syncthreads();
    _Float16* o = xT + (size_t)b * FF * NN + n0;
    for (int idx = threadIdx.x; idx < 64 * FF; idx += 256) {
        int f = idx >> 6, nl = idx & 63;               // consecutive t -> consecutive n
        o[(size_t)f * NN + nl] = tile[f][nl];
    }
}

// ---------------- kernel 5: attn_map + attn_feat (WMMA) ----------------
__global__ __launch_bounds__(256) void gat_main(const _Float16* __restrict__ xT,
                                                const float* __restrict__ k,
                                                const float* __restrict__ q,
                                                const float* __restrict__ qmax,
                                                const float* __restrict__ invz,
                                                float* __restrict__ feat,
                                                float* __restrict__ amap) {
    // double-buffered raw-e tile [buf][i][j], padded pitch 72 halves (144 B)
    __shared__ alignas(16) _Float16 eL[2][32][72];
    __shared__ float kS[32], mS[32], izS[32];

    const int b  = blockIdx.y;
    const int i0 = blockIdx.x * 32;                    // 32 i-rows per block
    const int t  = threadIdx.x;

    if (t < 32) {
        float kv = k[(size_t)b * NN + i0 + t];
        kS[t]  = kv;
        mS[t]  = lrelu(kv + qmax[b]);
        izS[t] = invz[(size_t)b * NN + i0 + t];
    }
    __syncthreads();

    const int lane = t & 31, wave = t >> 5;
    const int l = lane & 15, hi = lane >> 4;
    const int f0 = wave * 16;                          // wave owns f-columns [f0, f0+16)
    const int r0 = t >> 5;                             // phase-1 base i-row (rows r0+8*rr)
    const int jl = t & 31;                             // phase-1 j within half-chunk

    // hoist the per-thread row constants out of LDS
    float kr[4], mr[4], zr[4];
#pragma unroll
    for (int rr = 0; rr < 4; ++rr) {
        kr[rr] = kS[r0 + 8 * rr];
        mr[rr] = mS[r0 + 8 * rr];
        zr[rr] = izS[r0 + 8 * rr];
    }
    const float iz0 = izS[l];                          // C-tile 0 scaling (n = i0+l)
    const float iz1 = izS[16 + l];                     // C-tile 1 scaling (n = i0+16+l)

    const float*    qb    = q + (size_t)b * NN;
    float*          am    = amap + ((size_t)b * NN + i0 + r0) * NN;      // row r0
    const _Float16* aBase = xT + ((size_t)b * FF + f0 + l) * NN;         // row m = f0+l

    v8f c0 = {};
    v8f c1 = {};

    for (int jt = 0; jt < NN / 64; ++jt) {
        const int j0 = jt * 64;
        const int p  = jt & 1;

        // phase 1: fp32 softmax weights for a 32x64 tile; write normalized
        // attn_map non-temporally (coalesced 128B/wave), stash raw e as f16.
#pragma unroll
        for (int h = 0; h < 2; ++h) {
            int jc = h * 32 + jl;
            int j  = j0 + jc;
            float qv = qb[j];
#pragma unroll
            for (int rr = 0; rr < 4; ++rr) {
                float e = __expf(lrelu(kr[rr] + qv) - mr[rr]);
                __builtin_nontemporal_store(e * zr[rr], am + (size_t)(8 * rr) * NN + j);
                eL[p][r0 + 8 * rr][jc] = (_Float16)e;
            }
        }
        __syncthreads();   // single barrier: double-buffering covers WAR on eL

        // phase 2: four v_wmma_f32_16x16x32_f16 per wave; each A fragment
        // (global) feeds two N-tiles (B from LDS).
#pragma unroll
        for (int h = 0; h < 2; ++h) {
            const _Float16* ap = aBase + j0 + h * 32;

            // A fragment (16x32, M = f): K chunk at 16*hi
            v8h a0 = *(const v8h*)(ap + 16 * hi);
            v8h a1 = *(const v8h*)(ap + 16 * hi + 8);
            v16h A = __builtin_shufflevector(a0, a1, 0, 1, 2, 3, 4, 5, 6, 7,
                                             8, 9, 10, 11, 12, 13, 14, 15);

            // B fragments (32x16, N = i): columns l and l+16, K chunk at 16*hi
            const _Float16* bp0 = &eL[p][l][h * 32 + 16 * hi];
            const _Float16* bp1 = &eL[p][16 + l][h * 32 + 16 * hi];
            v8h b00 = *(const v8h*)(bp0);
            v8h b01 = *(const v8h*)(bp0 + 8);
            v8h b10 = *(const v8h*)(bp1);
            v8h b11 = *(const v8h*)(bp1 + 8);
            v16h B0 = __builtin_shufflevector(b00, b01, 0, 1, 2, 3, 4, 5, 6, 7,
                                              8, 9, 10, 11, 12, 13, 14, 15);
            v16h B1 = __builtin_shufflevector(b10, b11, 0, 1, 2, 3, 4, 5, 6, 7,
                                              8, 9, 10, 11, 12, 13, 14, 15);

            c0 = __builtin_amdgcn_wmma_f32_16x16x32_f16(
                     false, A, false, B0, (short)0, c0, false, false);
            c1 = __builtin_amdgcn_wmma_f32_16x16x32_f16(
                     false, A, false, B1, (short)0, c1, false, false);
        }

        // pull next iteration's 128B A-line toward the WGP
        if (jt + 1 < NN / 64)
            __builtin_prefetch(aBase + j0 + 64, 0, 3);
    }

    // Normalize by invZ of column i (N dim -> lane-uniform scalar) and store.
    // C layout: elem r -> (m = f0 + r + 8*hi, n = i0 [+16] + l); lanes 0-15
    // give 64B-contiguous segments along i.
#pragma unroll
    for (int r = 0; r < 8; ++r) { c0[r] *= iz0; c1[r] *= iz1; }

    float* fb = feat + (size_t)b * FF * NN;
#pragma unroll
    for (int r = 0; r < 8; ++r) {
        float* rowp = fb + (size_t)(f0 + r + 8 * hi) * NN + i0 + l;
        __builtin_nontemporal_store(c0[r], rowp);
        __builtin_nontemporal_store(c1[r], rowp + 16);
    }
}

// ---------------- host launcher ----------------
extern "C" void kernel_launch(void* const* d_in, const int* in_sizes, int n_in,
                              void* d_out, int out_size, void* d_ws, size_t ws_size,
                              hipStream_t stream) {
    (void)in_sizes; (void)n_in; (void)out_size; (void)ws_size;
    const float* x  = (const float*)d_in[0];
    const float* wk = (const float*)d_in[1];
    const float* wq = (const float*)d_in[2];

    char* ws = (char*)d_ws;
    float*    k    = (float*)(ws);                                   // 512 KB
    float*    q    = (float*)(ws + 512 * 1024);                      // 512 KB
    float*    qmax = (float*)(ws + 1024 * 1024);                     // 1 KB slot
    float*    invz = (float*)(ws + 1024 * 1024 + 1024);              // 512 KB
    _Float16* xT   = (_Float16*)(ws + 1024 * 1024 + 1024 + 512 * 1024); // 8 MB

    float* feat = (float*)d_out;                       // (B, F, N)
    float* amap = feat + (size_t)BB * FF * NN;         // (B, N, N)

    kq_kernel  <<<(BB * NN) / 256, 256, 0, stream>>>(x, wk, wq, k, q);
    qmax_kernel<<<BB, 256, 0, stream>>>(q, qmax);
    invz_kernel<<<(BB * NN) / 8, 256, 0, stream>>>(k, q, qmax, invz);
    xT_kernel  <<<dim3(NN / 64, BB), 256, 0, stream>>>(x, xT);
    gat_main   <<<dim3(NN / 32, BB), 256, 0, stream>>>(xT, k, q, qmax, invz, feat, amap);
}
